// Model_29454885716719
// MI455X (gfx1250) — matrix operand order repaired and checked by reading, not compiled, not run on hardware
//
#include <hip/hip_runtime.h>

typedef _Float16 h16;
typedef __attribute__((ext_vector_type(16))) _Float16 v16h;
typedef __attribute__((ext_vector_type(8)))  _Float16 v8h;
typedef __attribute__((ext_vector_type(8)))  float    v8f;

#define NB   64
#define LSEQ 8192
#define W1O  4096
#define W2O  2048
#define SEQ  1024
#define DF   20
#define DP   32

// ---------- helpers -------------------------------------------------------

__device__ __forceinline__ v16h load_pair16(const h16* p0, const h16* p1) {
  v8h lo = *(const v8h*)p0;
  v8h hi = *(const v8h*)p1;
  v16h r;
#pragma unroll
  for (int i = 0; i < 8; ++i) { r[i] = lo[i]; r[i + 8] = hi[i]; }
  return r;
}

__device__ __forceinline__ v8f wmma_f32(v16h a, v16h b, v8f c) {
  // D = A(16x32 f16) * B(32x16 f16) + C(16x16 f32)
  return __builtin_amdgcn_wmma_f32_16x16x32_f16(
      /*neg_a=*/false, a, /*neg_b=*/false, b,
      /*c_mod=*/(short)0, c, /*reuse_a=*/false, /*reuse_b=*/false);
}

// ---------- conv stack (scalar; ~0.8 GFLOP total, negligible) -------------

__global__ __launch_bounds__(256) void conv1_kernel(
    const int* __restrict__ inp, const float* __restrict__ w,
    const float* __restrict__ bias, float* __restrict__ out) {
  int t = blockIdx.x * 256 + threadIdx.x;
  int b = t >> 12, wp = t & 4095;
  int vals[4];
#pragma unroll
  for (int k = 0; k < 4; ++k) {
    int iw = 2 * wp - 1 + k;
    vals[k] = (iw >= 0 && iw < LSEQ) ? inp[(size_t)b * LSEQ + iw] : -1;
  }
  for (int h = 0; h < 4; ++h) {
    float acc[5];
#pragma unroll
    for (int c = 0; c < 5; ++c) acc[c] = bias[c];
#pragma unroll
    for (int kh = 0; kh < 4; ++kh) {
      int ih = h + kh - 1;
      if (ih < 0 || ih > 3) continue;
#pragma unroll
      for (int kw = 0; kw < 4; ++kw) {
        if (vals[kw] == ih) {
#pragma unroll
          for (int c = 0; c < 5; ++c) acc[c] += w[(kh * 4 + kw) * 5 + c];
        }
      }
    }
    float* o = out + (((size_t)(b * 4 + h)) * W1O + wp) * 5;
#pragma unroll
    for (int c = 0; c < 5; ++c) o[c] = acc[c];
  }
}

__global__ __launch_bounds__(256) void conv2_kernel(
    const float* __restrict__ in, const float* __restrict__ w,
    const float* __restrict__ bias, float* __restrict__ out) {
  int t = blockIdx.x * 256 + threadIdx.x;
  int b = t >> 11, wp = t & 2047;
  for (int h = 0; h < 4; ++h) {
    float acc[5];
#pragma unroll
    for (int c = 0; c < 5; ++c) acc[c] = bias[c];
    for (int kh = 0; kh < 4; ++kh) {
      int ih = h + kh - 1;
      if (ih < 0 || ih > 3) continue;
      for (int kw = 0; kw < 4; ++kw) {
        int iw = 2 * wp + kw - 1;
        if (iw < 0 || iw >= W1O) continue;
        const float* ip = in + (((size_t)(b * 4 + ih)) * W1O + iw) * 5;
        const float* wk = w + (kh * 4 + kw) * 25;
#pragma unroll
        for (int ci = 0; ci < 5; ++ci)
#pragma unroll
          for (int c = 0; c < 5; ++c) acc[c] += ip[ci] * wk[ci * 5 + c];
      }
    }
    float* o = out + (((size_t)(b * 4 + h)) * W2O + wp) * 5;
#pragma unroll
    for (int c = 0; c < 5; ++c) o[c] = acc[c];
  }
}

__global__ __launch_bounds__(256) void conv3_kernel(
    const float* __restrict__ in, const float* __restrict__ w,
    const float* __restrict__ bias, h16* __restrict__ x) {
  int t = blockIdx.x * 256 + threadIdx.x;
  int b = t >> 10, wp = t & 1023;
  h16* xr = x + ((size_t)(b * SEQ + wp)) * DP;
  for (int h = 0; h < 4; ++h) {
    float acc[5];
#pragma unroll
    for (int c = 0; c < 5; ++c) acc[c] = bias[c];
    for (int kh = 0; kh < 4; ++kh) {
      int ih = h + kh - 1;
      if (ih < 0 || ih > 3) continue;
      for (int kw = 0; kw < 4; ++kw) {
        int iw = 2 * wp + kw - 1;
        if (iw < 0 || iw >= W2O) continue;
        const float* ip = in + (((size_t)(b * 4 + ih)) * W2O + iw) * 5;
        const float* wk = w + (kh * 4 + kw) * 25;
#pragma unroll
        for (int ci = 0; ci < 5; ++ci)
#pragma unroll
          for (int c = 0; c < 5; ++c) acc[c] += ip[ci] * wk[ci * 5 + c];
      }
    }
#pragma unroll
    for (int c = 0; c < 5; ++c) xr[h * 5 + c] = (h16)acc[c];
  }
#pragma unroll
  for (int d = DF; d < DP; ++d) xr[d] = (h16)0.f;
}

// ---------- weight prep: [20,20] f32 -> padded/transposed [32,32] f16 -----

__global__ __launch_bounds__(256) void prep_w_kernel(
    const float* __restrict__ src, h16* __restrict__ dst) {
  int t = blockIdx.x * 256 + threadIdx.x;   // 4 blocks -> 1024 threads
  int f = t >> 5, d = t & 31;               // dst[f][d] = W[d][f]
  dst[f * 32 + d] = (f < DF && d < DF) ? (h16)src[d * DF + f] : (h16)0.f;
}

// ---------- Q/K/V projection via WMMA -------------------------------------
// x:[b][s][32] f16 row-major.  Wt:[f][d] 32x32 f16 (padded, transposed).
// Out: Q,K row-major [b][s][32]; Vt transposed [b][feat][s].
// Q/K C-fragments are transposed through a per-wave LDS slab (DS ops are
// in-order within a wave -> no barrier needed) so global stores are b128.

__global__ __launch_bounds__(256) void proj_kernel(
    const h16* __restrict__ x, const h16* __restrict__ Wq,
    const h16* __restrict__ Wk, const h16* __restrict__ Wv,
    h16* __restrict__ Q, h16* __restrict__ K, h16* __restrict__ Vt) {
  __shared__ __align__(16) h16 stage[8][16 * DP];   // 1KB per wave
  const int lane = threadIdx.x & 31;
  const int wv = threadIdx.x >> 5;
  const int gw = blockIdx.x * 8 + wv;
  const int b = gw >> 6;
  const int st = gw & 63;
  const int m = lane & 15;
  const int hh = lane >> 4;
  h16* lds = stage[wv];

  const h16* xrow = x + ((size_t)(b * SEQ + st * 16 + m)) * DP;
  v16h a = load_pair16(xrow + hh * 16, xrow + hh * 16 + 8);   // A: 16 rows x 32 feats

  const h16* Ws[3] = {Wq, Wk, Wv};
  h16* Outs[2] = {Q, K};
#pragma unroll
  for (int p = 0; p < 3; ++p) {
    const h16* wr0 = Ws[p] + (size_t)m * 32 + hh * 16;         // B: N=m, f 0..15
    const h16* wr1 = Ws[p] + (size_t)(16 + m) * 32 + hh * 16;  // B: N=m, f 16..31
    v8f c0 = {}, c1 = {};
    c0 = wmma_f32(a, load_pair16(wr0, wr0 + 8), c0);
    c1 = wmma_f32(a, load_pair16(wr1, wr1 + 8), c1);
    if (p < 2) {
      // transpose via LDS: write [s_local][f], read back packed rows
#pragma unroll
      for (int r = 0; r < 8; ++r) {
        lds[(8 * hh + r) * DP + m]      = (h16)c0[r];
        lds[(8 * hh + r) * DP + 16 + m] = (h16)c1[r];
      }
      v8h r0 = *(const v8h*)(lds + m * DP + hh * 16);
      v8h r1 = *(const v8h*)(lds + m * DP + hh * 16 + 8);
      h16* drow = Outs[p] + ((size_t)(b * SEQ + st * 16 + m)) * DP + hh * 16;
      *(v8h*)drow = r0;
      *(v8h*)(drow + 8) = r1;
    } else {
      // Vt rows are contiguous in s -> direct packed stores
      v8h pk0, pk1;
#pragma unroll
      for (int r = 0; r < 8; ++r) { pk0[r] = (h16)c0[r]; pk1[r] = (h16)c1[r]; }
      *(v8h*)(Vt + ((size_t)(b * DP + m)) * SEQ + st * 16 + hh * 8) = pk0;
      *(v8h*)(Vt + ((size_t)(b * DP + 16 + m)) * SEQ + st * 16 + hh * 8) = pk1;
    }
  }
}

// ---------- causal flash attention, one wave per (batch, 16-query tile) ---
// Computes S^T = K*Q^T so each lane owns one query column: online-softmax
// stats are per-lane scalars; cross-half reduction is a single shfl_xor(16).
// O^T = V^T * P accumulated in two 16x16 f32 WMMA fragments.

__global__ __launch_bounds__(256) void flash_kernel(
    const h16* __restrict__ Q, const h16* __restrict__ K,
    const h16* __restrict__ Vt, h16* __restrict__ xout) {
  const int lane = threadIdx.x & 31;
  const int gw = blockIdx.x * 8 + (threadIdx.x >> 5);
  const int b = gw >> 6;
  const int qt = gw & 63;
  const int m = lane & 15;
  const int hh = lane >> 4;
  const int qg = qt * 16 + m;                 // this lane's query
  const float scale = 0.2236067977499790f;    // 1/sqrt(20)

  // B-fragment of Q^T: lane N=query m, Kdim feats hh*16..hh*16+15 (contiguous)
  const h16* qrow = Q + ((size_t)(b * SEQ + qg)) * DP + hh * 16;
  v16h bq = load_pair16(qrow, qrow + 8);

  float mrun = -1e30f, lrun = 0.f;
  v8f o0 = {}, o1 = {};
  const int nkc = (qt * 16 + 15) / 32 + 1;    // causal: 32-key chunks needed

  for (int kc = 0; kc < nkc; ++kc) {
    // issue ALL fragment loads up front so they overlap WMMA + softmax VALU
    const h16* kb  = K + ((size_t)(b * SEQ + kc * 32)) * DP;
    const h16* kr0 = kb + (size_t)m * DP;
    const h16* kr1 = kb + (size_t)(16 + m) * DP;
    const h16* v0  = Vt + ((size_t)(b * DP + m)) * SEQ + kc * 32;
    const h16* v1  = Vt + ((size_t)(b * DP + 16 + m)) * SEQ + kc * 32;
    v16h ak0 = load_pair16(kr0 + hh * 16, kr0 + hh * 16 + 8);
    v16h ak1 = load_pair16(kr1 + hh * 16, kr1 + hh * 16 + 8);
    v16h av0 = load_pair16(v0 + hh * 16, v0 + hh * 16 + 8);
    v16h av1 = load_pair16(v1 + hh * 16, v1 + hh * 16 + 8);
    if (kc + 1 < nkc) {
      __builtin_prefetch(kb + (size_t)32 * DP, 0, 1);
      __builtin_prefetch(v0 + 32, 0, 1);
    }

    v8f z = {};
    v8f t1 = wmma_f32(ak0, bq, z);            // S^T keys kc*32+0..15
    v8f t2 = wmma_f32(ak1, bq, z);            // S^T keys kc*32+16..31

    float p1[8], p2[8];
    float tmax = -1e30f;
    const int kb0 = kc * 32 + hh * 8;
#pragma unroll
    for (int r = 0; r < 8; ++r) {
      float s1 = (kb0 + r      <= qg) ? t1[r] * scale : -1e30f;
      float s2 = (kb0 + 16 + r <= qg) ? t2[r] * scale : -1e30f;
      p1[r] = s1; p2[r] = s2;
      tmax = fmaxf(tmax, fmaxf(s1, s2));
    }
    tmax = fmaxf(tmax, __shfl_xor(tmax, 16, 32));
    float mnew = fmaxf(mrun, tmax);
    float corr = __expf(mrun - mnew);
    float psum = 0.f;
#pragma unroll
    for (int r = 0; r < 8; ++r) {
      p1[r] = __expf(p1[r] - mnew);
      p2[r] = __expf(p2[r] - mnew);
      psum += p1[r] + p2[r];
    }
    psum += __shfl_xor(psum, 16, 32);
    lrun = lrun * corr + psum;
    mrun = mnew;
#pragma unroll
    for (int i = 0; i < 8; ++i) { o0[i] *= corr; o1[i] *= corr; }

    // Build B-fragment of P (32 keys x 16 queries) from the two S^T frags
    // via partner half-wave exchanges -- no LDS transpose needed.
    v16h bp;
#pragma unroll
    for (int r = 0; r < 8; ++r) {
      float q1 = __shfl_xor(p1[r], 16, 32);
      float q2 = __shfl_xor(p2[r], 16, 32);
      bp[r]     = (h16)(hh ? q2    : p1[r]);  // Kdim keys hh*16 + 0..7
      bp[8 + r] = (h16)(hh ? p2[r] : q1);     // Kdim keys hh*16 + 8..15
    }

    o0 = wmma_f32(av0, bp, o0);               // O^T feats 0..15
    o1 = wmma_f32(av1, bp, o1);               // O^T feats 16..31
  }

  // Epilogue: lane-pair exchange so each lane writes 16 contiguous halves
  // (two global_store_b128) instead of 16 scattered b16 stores.
  const float linv = 1.f / lrun;
  float myv0[8], myv1[8];
#pragma unroll
  for (int r = 0; r < 8; ++r) { myv0[r] = o0[r] * linv; myv1[r] = o1[r] * linv; }
  v8h s0, s1;
#pragma unroll
  for (int r = 0; r < 8; ++r) {
    float e0 = __shfl_xor(myv0[r], 16, 32);
    float e1 = __shfl_xor(myv1[r], 16, 32);
    s0[r] = (h16)(hh ? e1      : myv0[r]);  // hh=0: feats 0-7 ; hh=1: feats 16-23
    s1[r] = (h16)(hh ? myv1[r] : e0);       // hh=0: feats 8-15; hh=1: feats 24-31
  }
  h16* obase = xout + ((size_t)(b * SEQ + qg)) * DP + hh * 16;
  *(v8h*)obase = s0;
  *(v8h*)(obase + 8) = s1;
}

// ---------- dense head ----------------------------------------------------

__global__ __launch_bounds__(256) void dense_kernel(
    const h16* __restrict__ xf, const float* __restrict__ w1,
    const float* __restrict__ b1, const float* __restrict__ w2,
    const float* __restrict__ b2, const float* __restrict__ w3,
    const float* __restrict__ b3, float* __restrict__ out) {
  const int b = blockIdx.x, tid = threadIdx.x;
  __shared__ float sums[10];
  if (tid < 10) sums[tid] = 0.f;
  __syncthreads();
  float acc[10];
#pragma unroll
  for (int j = 0; j < 10; ++j) acc[j] = 0.f;
  for (int idx = tid; idx < SEQ * DF; idx += 256) {
    int s = idx / DF, d = idx - s * DF;
    float xv = (float)xf[((size_t)(b * SEQ + s)) * DP + d];
    const float* wr = w1 + (size_t)idx * 10;
#pragma unroll
    for (int j = 0; j < 10; ++j) acc[j] += xv * wr[j];
  }
#pragma unroll
  for (int j = 0; j < 10; ++j) atomicAdd(&sums[j], acc[j]);
  __syncthreads();
  if (tid == 0) {
    float y1[10], y2[10], y3[10];
#pragma unroll
    for (int j = 0; j < 10; ++j) y1[j] = sums[j] + b1[j];
#pragma unroll
    for (int j = 0; j < 10; ++j) {
      float a = b2[j];
      for (int i = 0; i < 10; ++i) a += y1[i] * w2[i * 10 + j];
      y2[j] = a;
    }
#pragma unroll
    for (int j = 0; j < 10; ++j) {
      float a = b3[j];
      for (int i = 0; i < 10; ++i) a += y2[i] * w3[i * 10 + j];
      y3[j] = a;
    }
    for (int j = 0; j < 10; ++j) out[b * 10 + j] = y3[j];
  }
}

// ---------- launch --------------------------------------------------------

extern "C" void kernel_launch(void* const* d_in, const int* in_sizes, int n_in,
                              void* d_out, int out_size, void* d_ws, size_t ws_size,
                              hipStream_t stream) {
  const int*   inp = (const int*)d_in[0];
  const float* cw1 = (const float*)d_in[1];  const float* cb1 = (const float*)d_in[2];
  const float* cw2 = (const float*)d_in[3];  const float* cb2 = (const float*)d_in[4];
  const float* cw3 = (const float*)d_in[5];  const float* cb3 = (const float*)d_in[6];
  const float* dw1 = (const float*)d_in[16]; const float* db1 = (const float*)d_in[17];
  const float* dw2 = (const float*)d_in[18]; const float* db2 = (const float*)d_in[19];
  const float* dw3 = (const float*)d_in[20]; const float* db3 = (const float*)d_in[21];

  char* p = (char*)d_ws;
  // conv scratch (retired after conv3) is reused for Q/K/Vt/xB
  float* out1 = (float*)p;                                    // 20,971,520 B
  float* out2 = (float*)(p + 20971520u);                      // 10,485,760 B
  h16*   xA   = (h16*)(p + 20971520u + 10485760u);            //  4,194,304 B
  h16*   Wp   = (h16*)(p + 20971520u + 10485760u + 4194304u); //     18,432 B
  h16*   Qb   = (h16*)p;                                      // alias out1[0:4M)
  h16*   Kb   = (h16*)(p + 4194304u);                         // alias out1[4:8M)
  h16*   Vt   = (h16*)(p + 8388608u);                         // alias out1[8:12M)
  h16*   xB   = (h16*)(p + 12582912u);                        // alias out1[12:16M)

  conv1_kernel<<<1024, 256, 0, stream>>>(inp, cw1, cb1, out1);
  conv2_kernel<<<512, 256, 0, stream>>>(out1, cw2, cb2, out2);
  conv3_kernel<<<256, 256, 0, stream>>>(out2, cw3, cb3, xA);

  for (int i = 0; i < 9; ++i)
    prep_w_kernel<<<4, 256, 0, stream>>>((const float*)d_in[7 + i], Wp + i * 1024);

  h16* xc = xA;
  h16* xn = xB;
  for (int blk = 0; blk < 3; ++blk) {
    const h16* Wk = Wp + (blk * 3 + 0) * 1024;
    const h16* Wq = Wp + (blk * 3 + 1) * 1024;
    const h16* Wv = Wp + (blk * 3 + 2) * 1024;
    proj_kernel<<<512, 256, 0, stream>>>(xc, Wq, Wk, Wv, Qb, Kb, Vt);
    flash_kernel<<<512, 256, 0, stream>>>(Qb, Kb, Vt, xn);
    h16* t = xc; xc = xn; xn = t;
  }

  dense_kernel<<<64, 256, 0, stream>>>(xc, dw1, db1, dw2, db2, dw3, db3, (float*)d_out);
}
